// Attention_14190571946482
// MI455X (gfx1250) — matrix operand order repaired and checked by reading, not compiled, not run on hardware
//
#include <hip/hip_runtime.h>
#include <hip/hip_bf16.h>

typedef __attribute__((ext_vector_type(16))) __bf16 v16bf;
typedef __attribute__((ext_vector_type(8)))  float  v8f;

#define BATCH 4
#define SEQ   2048
#define DIM   512
#define BSROWS (BATCH * SEQ)
#define TQ 32
#define TK 32
#define NEGBIG (-1e12f)
#define INV_SQRT_D 0.04419417382415922f   // 1/sqrt(512)

union FragU { v16bf v; uint4 q[2]; };

// A-matrix 16x32 bf16 fragment (ISA 7.12.2): lane m = lane&15, half = lane>>4.
// VGPRs 0-3: K = half*8 .. half*8+7 ; VGPRs 4-7: K = 16+half*8 .. +7
__device__ __forceinline__ v16bf load_a_frag(const __bf16* base, int ld) {
    const int lane = threadIdx.x & 31;
    const int row  = lane & 15;
    const int half = lane >> 4;
    const __bf16* p = base + (size_t)row * ld + half * 8;
    FragU f;
    f.q[0] = *(const uint4*)(p);
    f.q[1] = *(const uint4*)(p + 16);
    return f.v;
}

// B-matrix 32x16 bf16 fragment: lane n = lane&15, half = lane>>4.
// Per lane: K = half*8 .. half*8+7, then K = 16+half*8 .. +7 (matches A).
__device__ __forceinline__ v16bf load_b_frag(const __bf16* base, int ld) {
    const int lane = threadIdx.x & 31;
    const int col  = lane & 15;
    const int half = lane >> 4;
    const __bf16* p = base + (size_t)col * ld + half * 8;
    FragU f;
    f.q[0] = *(const uint4*)(p);
    f.q[1] = *(const uint4*)(p + 16);
    return f.v;
}

__device__ __forceinline__ v8f wmma_bf16(v16bf a, v16bf b, v8f c) {
    return __builtin_amdgcn_wmma_f32_16x16x32_bf16(
        false, a, false, b, (short)0, c, false, false);
}

// Async global -> LDS copy, 16B per lane. ISA 10.2: generic LDS address
// truncates to the LDS byte offset (addr[31:0]).
__device__ __forceinline__ void async_copy_b128(const void* gptr, void* lptr) {
    unsigned lds = (unsigned)(size_t)lptr;
    asm volatile("global_load_async_to_lds_b128 %0, %1, off"
                 :: "v"(lds), "v"(gptr) : "memory");
}
__device__ __forceinline__ void wait_async_le16() {
    asm volatile("s_wait_asynccnt 16" ::: "memory");
}
__device__ __forceinline__ void wait_async_0() {
    asm volatile("s_wait_asynccnt 0" ::: "memory");
}

// ---------------------------------------------------------------- fp32 -> bf16
__global__ __launch_bounds__(256) void cvt_kernel(const float* __restrict__ x,
                                                  __bf16* __restrict__ y, int n) {
    int i = blockIdx.x * 256 + threadIdx.x;
    if (i < n) y[i] = (__bf16)x[i];
}

// -------------------------------------------------- W [D][U] fp32 -> Wt [U][D] bf16
__global__ __launch_bounds__(256) void wtrans_kernel(const float* __restrict__ w,
                                                     __bf16* __restrict__ wt) {
    int i = blockIdx.x * 256 + threadIdx.x;
    if (i < DIM * DIM) {
        int d = i / DIM, u = i % DIM;
        wt[(size_t)u * DIM + d] = (__bf16)w[i];
    }
}

// -------------------------------------------------------------- projections
__global__ __launch_bounds__(256) void proj_kernel(
    const __bf16* __restrict__ Xb,
    const __bf16* __restrict__ WtQ, const __bf16* __restrict__ WtK,
    const __bf16* __restrict__ WtV,
    __bf16* __restrict__ qb, __bf16* __restrict__ kb, __bf16* __restrict__ vT) {

    const int mt = blockIdx.x, nt = blockIdx.y, which = blockIdx.z;
    const int tid = threadIdx.x, wave = tid >> 5, lane = tid & 31;
    const int mi = wave & 3;
    const int nj = wave >> 2;
    const int half = lane >> 4, ln = lane & 15;

    const __bf16* Wt = (which == 0) ? WtQ : (which == 1) ? WtK : WtV;
    const __bf16* arow = Xb + (size_t)(mt * 64 + mi * 16) * DIM;
    const __bf16* b0   = Wt + (size_t)(nt * 64 + nj * 32) * DIM;
    const __bf16* b1   = b0 + (size_t)16 * DIM;

    v8f acc0 = {}, acc1 = {};
    for (int kk = 0; kk < DIM; kk += 32) {
        __builtin_prefetch(arow + kk + 128, 0, 1);
        v16bf a  = load_a_frag(arow + kk, DIM);
        v16bf f0 = load_b_frag(b0 + kk, DIM);
        v16bf f1 = load_b_frag(b1 + kk, DIM);
        acc0 = wmma_bf16(a, f0, acc0);
        acc1 = wmma_bf16(a, f1, acc1);
    }

    const float scale = (which == 0) ? INV_SQRT_D : 1.0f;
#pragma unroll
    for (int r = 0; r < 8; ++r) {
        int row = mt * 64 + mi * 16 + 8 * half + r;
        int u0  = nt * 64 + nj * 32 + ln;
        int u1  = u0 + 16;
        float x0 = acc0[r] * scale, x1 = acc1[r] * scale;
        if (which == 2) {
            int b = row / SEQ, s = row % SEQ;
            vT[((size_t)b * DIM + u0) * SEQ + s] = (__bf16)x0;
            vT[((size_t)b * DIM + u1) * SEQ + s] = (__bf16)x1;
        } else {
            __bf16* dst = (which == 0) ? qb : kb;
            dst[(size_t)row * DIM + u0] = (__bf16)x0;
            dst[(size_t)row * DIM + u1] = (__bf16)x1;
        }
    }
}

// ------------------------------------------------------------ flash attention
// grid: (SEQ/TQ, BATCH). 8 waves: 2 m-tiles x 4 u-quarters. K/V tiles double-
// buffered in LDS via async copies; q tile resident in LDS for all iterations.
__global__ __launch_bounds__(256) void flash_kernel(
    const __bf16* __restrict__ qb, const __bf16* __restrict__ kb,
    const __bf16* __restrict__ vT, float* __restrict__ out) {

    const int qt = blockIdx.x, b = blockIdx.y;
    const int qbase = qt * TQ;
    const int tid = threadIdx.x, wave = tid >> 5, lane = tid & 31;
    const int mi = wave & 1;
    const int uj = wave >> 1;
    const int half = lane >> 4, ln = lane & 15;

    __shared__ __align__(16) __bf16 sQ[TQ][DIM];        // 32 KB
    __shared__ __align__(16) __bf16 sK[2][TK][DIM];     // 2 x 32 KB
    __shared__ __align__(16) __bf16 sV[2][DIM][TK];     // 2 x 32 KB
    __shared__ __align__(16) float  sS[TQ][TK];
    __shared__ __align__(16) __bf16 sP[TQ][TK];
    __shared__ float sM[TQ], sL[TQ], sC[TQ];

    v8f o[8];
#pragma unroll
    for (int t = 0; t < 8; ++t) o[t] = (v8f){};

    if (tid < TQ) { sM[tid] = -__builtin_inff(); sL[tid] = 0.0f; }

    // ---- stage q tile: 32x512 bf16 = 2048 16B chunks, 8 per thread (8 async instr/wave)
#pragma unroll
    for (int i = 0; i < 8; ++i) {
        int c = tid + i * 256;
        int row = c >> 6, col = (c & 63) * 8;
        async_copy_b128(qb + (size_t)(b * SEQ + qbase + row) * DIM + col,
                        &sQ[row][col]);
    }

    const int nkt = (qt == 0) ? (SEQ / TK) : (qt + 1);

    // ---- prologue: issue K/V tile 0 (16 async instr/wave)
    {
        const int kbase = 0;
#pragma unroll
        for (int i = 0; i < 8; ++i) {
            int c = tid + i * 256;
            int row = c >> 6, col = (c & 63) * 8;
            async_copy_b128(kb + (size_t)(b * SEQ + kbase + row) * DIM + col,
                            &sK[0][row][col]);
        }
#pragma unroll
        for (int i = 0; i < 8; ++i) {
            int c = tid + i * 256;
            int row = c >> 2, col = (c & 3) * 8;
            async_copy_b128(vT + ((size_t)b * DIM + row) * SEQ + kbase + col,
                            &sV[0][row][col]);
        }
    }

    for (int kt = 0; kt < nkt; ++kt) {
        const int cur = kt & 1, nxt = cur ^ 1;
        const int kbase = kt * TK;

        // issue next tile into the other buffer (safe: loop-end barrier of the
        // previous iteration ordered all compute on that buffer before now)
        if (kt + 1 < nkt) {
            const int nb = (kt + 1) * TK;
#pragma unroll
            for (int i = 0; i < 8; ++i) {
                int c = tid + i * 256;
                int row = c >> 6, col = (c & 63) * 8;
                async_copy_b128(kb + (size_t)(b * SEQ + nb + row) * DIM + col,
                                &sK[nxt][row][col]);
            }
#pragma unroll
            for (int i = 0; i < 8; ++i) {
                int c = tid + i * 256;
                int row = c >> 2, col = (c & 3) * 8;
                async_copy_b128(vT + ((size_t)b * DIM + row) * SEQ + nb + col,
                                &sV[nxt][row][col]);
            }
            wait_async_le16();   // only the just-issued batch may remain in flight
        } else {
            wait_async_0();
        }
        __syncthreads();         // all waves' slices of buffer `cur` visible

        // ---- score tile from LDS (waves uj<2 compute the 2x2 WMMA tiles)
        if (uj < 2) {
            const int nj = uj;
            v8f acc = {};
            for (int kk = 0; kk < DIM; kk += 32) {
                v16bf a = load_a_frag(&sQ[mi * 16][kk], DIM);
                v16bf f = load_b_frag(&sK[cur][nj * 16][kk], DIM);
                acc = wmma_bf16(a, f, acc);
            }
#pragma unroll
            for (int r = 0; r < 8; ++r) {
                int row = mi * 16 + 8 * half + r;
                int col = nj * 16 + ln;
                float s = acc[r];
                if (kbase + col >= qbase + row) s += NEGBIG;  // j >= i masked
                sS[row][col] = s;
            }
        }
        __syncthreads();

        // ---- online softmax: one thread per row
        if (tid < TQ) {
            const int row = tid;
            float mOld = sM[row];
            float tmax = -__builtin_inff();
            for (int c = 0; c < TK; ++c) tmax = fmaxf(tmax, sS[row][c]);
            float mNew = fmaxf(mOld, tmax);
            float corr = __expf(mOld - mNew);
            float sum = 0.0f;
            for (int c = 0; c < TK; ++c) {
                float p = __expf(sS[row][c] - mNew);
                sum += p;
                sP[row][c] = (__bf16)p;
            }
            sM[row] = mNew;
            sL[row] = sL[row] * corr + sum;
            sC[row] = corr;
        }
        __syncthreads();

        // ---- rescale accumulators, then O += P @ V (V from LDS)
#pragma unroll
        for (int r = 0; r < 8; ++r) {
            float corr = sC[mi * 16 + 8 * half + r];
#pragma unroll
            for (int t = 0; t < 8; ++t) o[t][r] *= corr;
        }
        v16bf pa = load_a_frag(&sP[mi * 16][0], TK);
#pragma unroll
        for (int t = 0; t < 8; ++t) {
            v16bf f = load_b_frag(&sV[cur][uj * 128 + t * 16][0], TK);
            o[t] = wmma_bf16(pa, f, o[t]);
        }
        __syncthreads();
    }

    // ---- normalize by row sums and store fp32
#pragma unroll
    for (int r = 0; r < 8; ++r) {
        int row = mi * 16 + 8 * half + r;
        float invl = 1.0f / sL[row];
        size_t grow = (size_t)(b * SEQ + qbase + row);
#pragma unroll
        for (int t = 0; t < 8; ++t) {
            int u = uj * 128 + t * 16 + ln;
            out[grow * DIM + u] = o[t][r] * invl;
        }
    }
}

extern "C" void kernel_launch(void* const* d_in, const int* in_sizes, int n_in,
                              void* d_out, int out_size, void* d_ws, size_t ws_size,
                              hipStream_t stream) {
    (void)in_sizes; (void)n_in; (void)out_size; (void)ws_size;
    const float* X  = (const float*)d_in[0];
    const float* Wq = (const float*)d_in[1];  // q projection
    const float* Wv = (const float*)d_in[2];  // k projection (reference naming quirk)
    const float* Wk = (const float*)d_in[3];  // v projection
    float* out = (float*)d_out;

    char* ws = (char*)d_ws;
    __bf16* Xb  = (__bf16*)ws; ws += (size_t)BSROWS * DIM * 2;
    __bf16* qb  = (__bf16*)ws; ws += (size_t)BSROWS * DIM * 2;
    __bf16* kb  = (__bf16*)ws; ws += (size_t)BSROWS * DIM * 2;
    __bf16* vT  = (__bf16*)ws; ws += (size_t)BSROWS * DIM * 2;
    __bf16* WtQ = (__bf16*)ws; ws += (size_t)DIM * DIM * 2;
    __bf16* WtK = (__bf16*)ws; ws += (size_t)DIM * DIM * 2;
    __bf16* WtV = (__bf16*)ws; ws += (size_t)DIM * DIM * 2;

    const int nX = BSROWS * DIM;
    cvt_kernel<<<(nX + 255) / 256, 256, 0, stream>>>(X, Xb, nX);
    const int nW = DIM * DIM;
    wtrans_kernel<<<(nW + 255) / 256, 256, 0, stream>>>(Wq, WtQ);
    wtrans_kernel<<<(nW + 255) / 256, 256, 0, stream>>>(Wv, WtK);
    wtrans_kernel<<<(nW + 255) / 256, 256, 0, stream>>>(Wk, WtV);

    proj_kernel<<<dim3(BSROWS / 64, DIM / 64, 3), 256, 0, stream>>>(
        Xb, WtQ, WtK, WtV, qb, kb, vT);

    flash_kernel<<<dim3(SEQ / TQ, BATCH), 256, 0, stream>>>(qb, kb, vT, out);
}
